// EdgeEncoder_71889162600811
// MI455X (gfx1250) — matrix operand order repaired and checked by reading, not compiled, not run on hardware
//
#include <hip/hip_runtime.h>
#include <hip/hip_fp16.h>
#include <math.h>

typedef _Float16 h16;
typedef __attribute__((ext_vector_type(16))) _Float16 v16h;
typedef __attribute__((ext_vector_type(8)))  _Float16 v8h;
typedef __attribute__((ext_vector_type(8)))  float    v8f;

#define NHEADS 4
#define LEAKY  0.2f
#define BNEPS  1e-5f

// ---------------------------------------------------------------- utilities
__global__ void k_cast_f16(const float* __restrict__ src, h16* __restrict__ dst, int n) {
  int i = blockIdx.x * blockDim.x + threadIdx.x;
  if (i < n) dst[i] = (h16)src[i];
}

// weights: f32 [K,Nc] row-major -> f16 [Nc,K] (transposed, so WMMA B-fragments
// become contiguous 32-byte runs loadable with global_load_b128 pairs)
__global__ void k_cast_transpose(const float* __restrict__ src, h16* __restrict__ dst,
                                 int K, int Nc) {
  int i = blockIdx.x * blockDim.x + threadIdx.x;
  if (i >= K * Nc) return;
  int k = i / Nc, n = i - k * Nc;
  dst[(size_t)n * K + k] = (h16)src[i];
}

// xi = x + ne[node_type], cast to f16
__global__ void k_xi_cast(const float* __restrict__ x, const float* __restrict__ ne,
                          const int* __restrict__ nt, h16* __restrict__ dst,
                          int N, int F) {
  int i = blockIdx.x * blockDim.x + threadIdx.x;
  if (i >= N * F) return;
  int n = i / F, f = i - n * F;
  dst[i] = (h16)(x[i] + ne[nt[n] * F + f]);
}

__global__ void k_init_maxden(float* __restrict__ amax, float* __restrict__ den, int n) {
  int i = blockIdx.x * blockDim.x + threadIdx.x;
  if (i < n) { amax[i] = -INFINITY; den[i] = 0.0f; }
}

__device__ inline void atomicMaxF(float* addr, float v) {
  // sign-split trick: works for mixed-sign values; init must be -inf
  if (v >= 0.0f) atomicMax((int*)addr, __float_as_int(v));
  else           atomicMin((unsigned int*)addr, __float_as_uint(v));
}

// ---------------------------------------------------------------- WMMA GEMM
// C[M,Nc] = A[M,K](f16,row-major) * B(from Bt: f16 [Nc,K] transposed) + bias.
// K%32==0, Nc%64==0. Block = 128 threads (4 waves). Each wave computes a
// 16(M) x 64(N) strip: one A fragment reused across 4 B fragments -> 4 WMMAs
// per k-step. grid = (ceil(M/64), Nc/64).
__global__ __launch_bounds__(128)
void k_wmma_gemm(const h16* __restrict__ A, const h16* __restrict__ Bt,
                 float* __restrict__ C, const float* __restrict__ bias,
                 int M, int K, int Nc) {
  const int lane = threadIdx.x & 31;
  const int wave = threadIdx.x >> 5;
  const int g    = lane >> 4;       // half-wave group
  const int mr   = lane & 15;       // A row / B col within 16x16 tile
  const int m0   = blockIdx.x * 64 + wave * 16;
  if (m0 >= M) return;              // wave-uniform: EXEC stays all-ones
  const int n0   = blockIdx.y * 64;

  v8f acc0 = {}, acc1 = {}, acc2 = {}, acc3 = {};
  // A: lane reads row (m0+mr), K-offsets {16g..16g+15} (contiguous)
  const h16* arow = A + (size_t)(m0 + mr) * K + 16 * g;
  // Bt: lane reads "column" n as contiguous run of 16 halves at k0+16g
  const h16* bbase = Bt + (size_t)(n0 + mr) * K + 16 * g;
  const size_t bstride = (size_t)16 * K;   // 16 columns apart

  for (int k0 = 0; k0 < K; k0 += 32) {
    union { v16h v; v8h h[2]; } au;
    au.h[0] = *(const v8h*)(arow + k0);
    au.h[1] = *(const v8h*)(arow + k0 + 8);
    v16h af = au.v;

    v16h b0 = *(const v16h*)(bbase + k0);
    v16h b1 = *(const v16h*)(bbase + k0 + bstride);
    v16h b2 = *(const v16h*)(bbase + k0 + 2 * bstride);
    v16h b3 = *(const v16h*)(bbase + k0 + 3 * bstride);

    acc0 = __builtin_amdgcn_wmma_f32_16x16x32_f16(false, af, false, b0, (short)0, acc0, false, false);
    acc1 = __builtin_amdgcn_wmma_f32_16x16x32_f16(false, af, false, b1, (short)0, acc1, false, false);
    acc2 = __builtin_amdgcn_wmma_f32_16x16x32_f16(false, af, false, b2, (short)0, acc2, false, false);
    acc3 = __builtin_amdgcn_wmma_f32_16x16x32_f16(false, af, false, b3, (short)0, acc3, false, false);
  }

  // C layout: lane -> col (n-subtile + mr), rows r + 8*g
  #pragma unroll
  for (int t = 0; t < 4; t++) {
    v8f acc = (t == 0) ? acc0 : (t == 1) ? acc1 : (t == 2) ? acc2 : acc3;
    int ncol = n0 + t * 16 + mr;
    float bv = bias ? bias[ncol] : 0.0f;
    #pragma unroll
    for (int r = 0; r < 8; r++) {
      int m = m0 + r + (g << 3);
      C[(size_t)m * Nc + ncol] = acc[r] + bv;
    }
  }
}

// ---------------------------------------------------------------- attention
// s[n,h] = sum_c h[n,h,c]*a_src[h,c]; d likewise
__global__ void k_attn_sd(const float* __restrict__ h, const float* __restrict__ asrc,
                          const float* __restrict__ adst, float* __restrict__ s,
                          float* __restrict__ d, int N, int C) {
  int i = blockIdx.x * blockDim.x + threadIdx.x;
  if (i >= N * NHEADS) return;
  int n = i >> 2, hh = i & 3;
  const float* hp = h + (size_t)n * NHEADS * C + hh * C;
  const float* as = asrc + hh * C;
  const float* ad = adst + hh * C;
  float ss = 0.0f, dd = 0.0f;
  for (int c = 0; c < C; c++) { float v = hp[c]; ss += v * as[c]; dd += v * ad[c]; }
  s[i] = ss; d[i] = dd;
}

// fold We @ a_edge -> Bedge[DE,4]; launch DE*NHEADS threads, 1 block
__global__ void k_bedge(const float* __restrict__ We, const float* __restrict__ ae,
                        float* __restrict__ B, int C) {
  int t = threadIdx.x;
  int dd = t >> 2, hh = t & 3;
  const float* wp = We + (size_t)dd * NHEADS * C + hh * C;
  const float* ap = ae + hh * C;
  float acc = 0.0f;
  for (int c = 0; c < C; c++) acc += wp[c] * ap[c];
  B[dd * NHEADS + hh] = acc;
}

// alpha[e,h] = leaky(s[src]+d[dst]+ (ea+ee[et]) @ Bedge); atomic segment max
__global__ void k_edge_alpha(const float* __restrict__ ea, const float* __restrict__ ee,
                             const int* __restrict__ et, const float* __restrict__ Bedge,
                             const float* __restrict__ s, const float* __restrict__ d,
                             const int* __restrict__ src, const int* __restrict__ dst,
                             float* __restrict__ alpha, float* __restrict__ amax,
                             int E, int DE) {
  int e = blockIdx.x * blockDim.x + threadIdx.x;
  if (e >= E) return;
  const float* eap = ea + (size_t)e * DE;
  const float* eep = ee + (size_t)et[e] * DE;
  float a0 = 0, a1 = 0, a2 = 0, a3 = 0;
  for (int q = 0; q < DE; q++) {
    float v = eap[q] + eep[q];
    const float* bp = Bedge + q * 4;
    a0 += v * bp[0]; a1 += v * bp[1]; a2 += v * bp[2]; a3 += v * bp[3];
  }
  float av[4] = {a0, a1, a2, a3};
  int si = src[e] * 4, di = dst[e] * 4;
  #pragma unroll
  for (int hh = 0; hh < 4; hh++) {
    float x = s[si + hh] + d[di + hh] + av[hh];
    x = x > 0.0f ? x : LEAKY * x;
    alpha[(size_t)e * 4 + hh] = x;
    atomicMaxF(&amax[di + hh], x);
  }
}

// ex = exp(alpha - amax[dst]) in place; den += ex
__global__ void k_edge_exp(float* __restrict__ alpha, const float* __restrict__ amax,
                           float* __restrict__ den, const int* __restrict__ dst, int E) {
  int i = blockIdx.x * blockDim.x + threadIdx.x;
  if (i >= E * 4) return;
  int e = i >> 2, hh = i & 3;
  int di = dst[e] * 4 + hh;
  float ex = __expf(alpha[i] - amax[di]);
  alpha[i] = ex;
  atomicAdd(&den[di], ex);
}

// layer-1 scatter (concat heads): out[dst, f] += h[src, f] * coef[head(f)]
__global__ __launch_bounds__(128)
void k_scatter_concat(const float* __restrict__ ex, const float* __restrict__ den,
                      const float* __restrict__ h, const int* __restrict__ src,
                      const int* __restrict__ dst, float* __restrict__ out,
                      int E, int C, int F) {
  int e = blockIdx.x;
  int f = threadIdx.x;            // 0..F-1 (F = NHEADS*C)
  int hh = f / C;
  int si = src[e], di = dst[e];
  float coef = ex[(size_t)e * 4 + hh] / (den[di * 4 + hh] + 1e-16f);
  atomicAdd(&out[(size_t)di * F + f], h[(size_t)si * F + f] * coef);
}

// layer-2 scatter (mean over heads): out[dst, c] += 0.25 * sum_h h[src,h,c]*coef[h]
__global__ __launch_bounds__(128)
void k_scatter_mean(const float* __restrict__ ex, const float* __restrict__ den,
                    const float* __restrict__ h, const int* __restrict__ src,
                    const int* __restrict__ dst, float* __restrict__ out,
                    int E, int C) {
  int e = blockIdx.x;
  int c = threadIdx.x;            // 0..C-1
  int si = src[e], di = dst[e];
  const float* hp = h + (size_t)si * NHEADS * C;
  float acc = 0.0f;
  #pragma unroll
  for (int hh = 0; hh < 4; hh++) {
    float coef = ex[(size_t)e * 4 + hh] / (den[di * 4 + hh] + 1e-16f);
    acc += hp[hh * C + c] * coef;
  }
  atomicAdd(&out[(size_t)di * C + c], 0.25f * acc);
}

// ---------------------------------------------------------------- batch norm
__global__ __launch_bounds__(256)
void k_bn_stats(const float* __restrict__ x, float* __restrict__ stats, int N, int F) {
  __shared__ float ss[256], sq[256];
  int f = blockIdx.x;
  float a = 0.0f, b = 0.0f;
  for (int n = threadIdx.x; n < N; n += 256) {
    float v = x[(size_t)n * F + f]; a += v; b += v * v;
  }
  ss[threadIdx.x] = a; sq[threadIdx.x] = b;
  __syncthreads();
  for (int o = 128; o > 0; o >>= 1) {
    if (threadIdx.x < o) { ss[threadIdx.x] += ss[threadIdx.x + o];
                           sq[threadIdx.x] += sq[threadIdx.x + o]; }
    __syncthreads();
  }
  if (threadIdx.x == 0) {
    float mu  = ss[0] / N;
    float var = sq[0] / N - mu * mu;
    stats[f] = mu; stats[F + f] = rsqrtf(var + BNEPS);
  }
}

__global__ void k_bn_apply(const float* __restrict__ x, const float* __restrict__ stats,
                           const float* __restrict__ gamma, const float* __restrict__ beta,
                           float* __restrict__ y, int total, int F) {
  int i = blockIdx.x * blockDim.x + threadIdx.x;
  if (i >= total) return;
  int f = i % F;
  y[i] = (x[i] - stats[f]) * stats[F + f] * gamma[f] + beta[f];
}

// ---------------------------------------------------------------- host
extern "C" void kernel_launch(void* const* d_in, const int* in_sizes, int n_in,
                              void* d_out, int out_size, void* d_ws, size_t ws_size,
                              hipStream_t stream) {
  const float* x     = (const float*)d_in[0];
  const int*   ei    = (const int*)d_in[1];
  const int*   nt    = (const int*)d_in[2];
  const float* eattr = (const float*)d_in[3];
  const int*   etype = (const int*)d_in[4];
  const float* ne1   = (const float*)d_in[5];
  const float* w1    = (const float*)d_in[6];
  const float* we1   = (const float*)d_in[7];
  const float* asrc1 = (const float*)d_in[8];
  const float* adst1 = (const float*)d_in[9];
  const float* aedge1= (const float*)d_in[10];
  const float* ee1   = (const float*)d_in[11];
  const float* wres1 = (const float*)d_in[12];
  const float* b1    = (const float*)d_in[13];
  const float* gamma = (const float*)d_in[14];
  const float* beta  = (const float*)d_in[15];
  const float* ne2   = (const float*)d_in[16];
  const float* w2    = (const float*)d_in[17];
  const float* we2   = (const float*)d_in[18];
  const float* asrc2 = (const float*)d_in[19];
  const float* adst2 = (const float*)d_in[20];
  const float* aedge2= (const float*)d_in[21];
  const float* ee2   = (const float*)d_in[22];
  const float* wres2 = (const float*)d_in[23];
  const float* b2    = (const float*)d_in[24];

  const int N  = in_sizes[2];          // nodes
  const int E  = in_sizes[4];          // edges
  const int F1 = in_sizes[0] / N;      // 32
  const int D1 = in_sizes[13];         // 128
  const int DE = in_sizes[3] / E;      // 64
  const int C1 = F1;                   // per-head ch layer1 (32)
  const int C2 = D1;                   // per-head ch layer2 (128)
  const int*   srcI = ei;
  const int*   dstI = ei + E;

  // ---- workspace layout (bytes)
  char* ws = (char*)d_ws;
  size_t off = 0;
  auto alloc = [&](size_t bytes) { size_t o = off; off += (bytes + 255) & ~size_t(255); return o; };
  float* alpha = (float*)(ws + alloc((size_t)E * 4 * 4));
  float* amax  = (float*)(ws + alloc((size_t)N * 4 * 4));
  float* den   = (float*)(ws + alloc((size_t)N * 4 * 4));
  float* sbuf  = (float*)(ws + alloc((size_t)N * 4 * 4));
  float* dbuf  = (float*)(ws + alloc((size_t)N * 4 * 4));
  float* Bedge = (float*)(ws + alloc((size_t)DE * 4 * 4));
  h16*   wAh   = (h16*)  (ws + alloc((size_t)D1 * NHEADS * D1 * 2));  // max weight (Bt)
  h16*   wRh   = (h16*)  (ws + alloc((size_t)D1 * D1 * 2));           // residual (Bt)
  float* stats = (float*)(ws + alloc((size_t)2 * D1 * 4));
  h16*   xiH   = (h16*)  (ws + alloc((size_t)N * D1 * 2));
  h16*   xH    = (h16*)  (ws + alloc((size_t)N * D1 * 2));
  float* hbuf  = (float*)(ws + alloc((size_t)N * NHEADS * D1 * 4)); // up to N*512
  float* aggr  = (float*)(ws + alloc((size_t)N * D1 * 4));
  float* hbn   = (float*)(ws + alloc((size_t)N * D1 * 4));
  (void)ws_size; (void)n_in; (void)out_size;
  float* outp = (float*)d_out;

  const int TB = 256;
  auto blks   = [&](long long n) { return (unsigned)((n + TB - 1) / TB); };
  auto mblks  = [&](int M) { return (unsigned)((M + 63) / 64); };

  // ================= layer 1 =================
  k_cast_transpose<<<blks(F1 * NHEADS * C1), TB, 0, stream>>>(w1, wAh, F1, NHEADS * C1);
  k_cast_transpose<<<blks(F1 * D1), TB, 0, stream>>>(wres1, wRh, F1, D1);
  k_xi_cast<<<blks((long long)N * F1), TB, 0, stream>>>(x, ne1, nt, xiH, N, F1);
  k_cast_f16<<<blks((long long)N * F1), TB, 0, stream>>>(x, xH, N * F1);

  // h1 = xi1 @ w1  [N,32]x[32,128]
  k_wmma_gemm<<<dim3(mblks(N), (NHEADS * C1) / 64), 128, 0, stream>>>(
      xiH, wAh, hbuf, nullptr, N, F1, NHEADS * C1);
  // aggr = x @ wres1 + b1  (residual seeds the aggregation buffer)
  k_wmma_gemm<<<dim3(mblks(N), D1 / 64), 128, 0, stream>>>(
      xH, wRh, aggr, b1, N, F1, D1);

  k_attn_sd<<<blks((long long)N * 4), TB, 0, stream>>>(hbuf, asrc1, adst1, sbuf, dbuf, N, C1);
  k_bedge<<<1, DE * NHEADS, 0, stream>>>(we1, aedge1, Bedge, C1);
  k_init_maxden<<<blks((long long)N * 4), TB, 0, stream>>>(amax, den, N * 4);
  k_edge_alpha<<<blks(E), TB, 0, stream>>>(eattr, ee1, etype, Bedge, sbuf, dbuf,
                                           srcI, dstI, alpha, amax, E, DE);
  k_edge_exp<<<blks((long long)E * 4), TB, 0, stream>>>(alpha, amax, den, dstI, E);
  k_scatter_concat<<<E, D1, 0, stream>>>(alpha, den, hbuf, srcI, dstI, aggr, E, C1, D1);

  // BatchNorm(128)
  k_bn_stats<<<D1, 256, 0, stream>>>(aggr, stats, N, D1);
  k_bn_apply<<<blks((long long)N * D1), TB, 0, stream>>>(aggr, stats, gamma, beta,
                                                         hbn, N * D1, D1);

  // ================= layer 2 =================
  k_cast_transpose<<<blks(D1 * NHEADS * C2), TB, 0, stream>>>(w2, wAh, D1, NHEADS * C2);
  k_cast_transpose<<<blks(D1 * D1), TB, 0, stream>>>(wres2, wRh, D1, D1);
  k_xi_cast<<<blks((long long)N * D1), TB, 0, stream>>>(hbn, ne2, nt, xiH, N, D1);
  k_cast_f16<<<blks((long long)N * D1), TB, 0, stream>>>(hbn, xH, N * D1);

  // h2 = xi2 @ w2  [N,128]x[128,512]
  k_wmma_gemm<<<dim3(mblks(N), (NHEADS * C2) / 64), 128, 0, stream>>>(
      xiH, wAh, hbuf, nullptr, N, D1, NHEADS * C2);
  // out = hbn @ wres2 + b2  (residual seeds output)
  k_wmma_gemm<<<dim3(mblks(N), D1 / 64), 128, 0, stream>>>(
      xH, wRh, outp, b2, N, D1, D1);

  k_attn_sd<<<blks((long long)N * 4), TB, 0, stream>>>(hbuf, asrc2, adst2, sbuf, dbuf, N, C2);
  k_bedge<<<1, DE * NHEADS, 0, stream>>>(we2, aedge2, Bedge, C2);
  k_init_maxden<<<blks((long long)N * 4), TB, 0, stream>>>(amax, den, N * 4);
  k_edge_alpha<<<blks(E), TB, 0, stream>>>(eattr, ee2, etype, Bedge, sbuf, dbuf,
                                           srcI, dstI, alpha, amax, E, DE);
  k_edge_exp<<<blks((long long)E * 4), TB, 0, stream>>>(alpha, amax, den, dstI, E);
  k_scatter_mean<<<E, C2, 0, stream>>>(alpha, den, hbuf, srcI, dstI, outp, E, C2);
}
